// GLABlock_14379550507703
// MI455X (gfx1250) — matrix operand order repaired and checked by reading, not compiled, not run on hardware
//
#include <hip/hip_runtime.h>

typedef __bf16 bf16;
typedef __attribute__((ext_vector_type(8)))  __bf16 v8bf;
typedef __attribute__((ext_vector_type(16))) __bf16 v16bf;
typedef __attribute__((ext_vector_type(8)))  float  v8f;
typedef __attribute__((ext_vector_type(4)))  unsigned int v4u;
typedef __attribute__((ext_vector_type(8)))  int v8i;
typedef __attribute__((ext_vector_type(4)))  int v4i;

static constexpr int B_ = 4, T_ = 2048, D_ = 2048, H_ = 16, DK_ = 64, DV_ = 128, CH_ = 64, LR_ = 16;
static constexpr int M_ = B_ * T_;                 // 8192 tokens
static constexpr float GATE_NORM_ = 16.0f, EPS_ = 1e-5f;

// ---- CDNA5 Tensor Data Mover availability -------------------------------
#if defined(__has_builtin)
#if __has_builtin(__builtin_amdgcn_tensor_load_to_lds)
#define USE_TDM 1
#endif
#endif
#ifndef USE_TDM
#define USE_TDM 0
#endif

#if USE_TDM
#warning "CDNA5 probe: tensor_load_to_lds builtin AVAILABLE - TDM staging path compiled"
#else
#warning "CDNA5 probe: tensor_load_to_lds builtin NOT available - sync staging fallback compiled"
#endif

__device__ __forceinline__ void wait_tensorcnt0() {
#if defined(__has_builtin) && __has_builtin(__builtin_amdgcn_s_wait_tensorcnt)
  __builtin_amdgcn_s_wait_tensorcnt(0);
#else
  asm volatile("s_wait_tensorcnt 0x0" ::: "memory");
#endif
}

#if USE_TDM
// Issue one TDM 2D tile load: tile (tile0 x tile1) elements of 2 bytes from a
// row-major tensor with row stride `stride0` (elements) into LDS at lds_addr.
// D# packing per CDNA5 ISA ch.8 (group0: 128b, group1: 256b, groups 2/3 zero).
__device__ __forceinline__ void tdm_load_2d_bf16(unsigned int lds_addr, const void* gaddr,
                                                 unsigned int dim0, unsigned int dim1,
                                                 unsigned int tile0, unsigned int tile1,
                                                 unsigned int stride0) {
  const unsigned long long ga = (unsigned long long)(__SIZE_TYPE__)gaddr;
  v4u g0;
  g0[0] = 1u;                                           // count=1, user mode
  g0[1] = lds_addr;                                     // lds_addr[31:0]
  g0[2] = (unsigned int)(ga & 0xffffffffu);             // global_addr[31:0]
  g0[3] = (unsigned int)((ga >> 32) & 0x1ffffffu) | (2u << 30);  // addr[56:32] | type=2
  v8i g1;
  g1[0] = (int)(1u << 16);                              // wg_mask=0, data_size=1 (2B)
  g1[1] = (int)((dim0 & 0xffffu) << 16);                // tensor_dim0[15:0] @ bits[63:48]
  g1[2] = (int)((dim0 >> 16) & 0xffffu) | (int)((dim1 & 0xffffu) << 16);
  g1[3] = (int)((dim1 >> 16) & 0xffffu) | (int)((tile0 & 0xffffu) << 16);
  g1[4] = (int)(tile1 & 0xffffu);                       // tile_dim1; tile_dim2=0
  g1[5] = (int)stride0;                                 // tensor_dim0_stride[31:0]
  g1[6] = 0;                                            // stride0[47:32], stride1[15:0]
  g1[7] = 0;
  v4i z4 = {0, 0, 0, 0};
#if __clang_major__ >= 23
  v8i z8 = {0, 0, 0, 0, 0, 0, 0, 0};
  __builtin_amdgcn_tensor_load_to_lds(g0, g1, z4, z4, z8, 0);
#else
  __builtin_amdgcn_tensor_load_to_lds(g0, g1, z4, z4, 0);
#endif
}
#endif

// ---------------------------------------------------------------- helpers
__device__ __forceinline__ float sigmoidf_(float x) { return 1.0f / (1.0f + __expf(-x)); }
__device__ __forceinline__ float log_sigmoidf_(float x) {
  return fminf(x, 0.0f) - log1pf(__expf(-fabsf(x)));
}
__device__ __forceinline__ float geluf_(float x) {
  return 0.5f * x * (1.0f + erff(x * 0.70710678118654752f));
}

__device__ __forceinline__ v8f wmma_bf16(v16bf a, v16bf b, v8f c) {
  return __builtin_amdgcn_wmma_f32_16x16x32_bf16(false, a, false, b, (short)0, c, false, false);
}

// A-fragment (16x32, 16-bit): lane half selects kb = half*16; 16 contiguous K per lane.
__device__ __forceinline__ v16bf load_a_frag(const bf16* base, int ld, int row, int kb) {
  const bf16* p = base + row * ld + kb;
  v8bf lo = *(const v8bf*)p;
  v8bf hi = *(const v8bf*)(p + 8);
  return __builtin_shufflevector(lo, hi, 0, 1, 2, 3, 4, 5, 6, 7, 8, 9, 10, 11, 12, 13, 14, 15);
}
// B-fragment (32x16, 16-bit), tile stored N-major [n][k]: 16 contiguous K per lane at kb = half*16.
__device__ __forceinline__ v16bf load_b_frag(const bf16* base, int ld, int col, int kb) {
  return *(const v16bf*)(base + col * ld + kb);
}

// ---------------------------------------------------------------- weight f32[K][N] -> bf16[N][K]
__global__ __launch_bounds__(256)
void cvt_transpose(const float* __restrict__ in, bf16* __restrict__ out, int K, int N) {
  __shared__ float tile[32][33];
  const int n0 = blockIdx.x * 32, k0 = blockIdx.y * 32;
  const int tx = threadIdx.x & 31, ty = threadIdx.x >> 5;  // 8 rows / pass
  for (int i = 0; i < 32; i += 8)
    tile[ty + i][tx] = in[(size_t)(k0 + ty + i) * N + n0 + tx];
  __syncthreads();
  for (int i = 0; i < 32; i += 8)
    out[(size_t)(n0 + ty + i) * K + k0 + tx] = (bf16)tile[tx][ty + i];
}

// ---------------------------------------------------------------- LayerNorm -> bf16
__global__ __launch_bounds__(256)
void layernorm_to_bf16(const float* __restrict__ x, const float* __restrict__ w,
                       const float* __restrict__ b, bf16* __restrict__ out) {
  __shared__ float red[256];
  const int t = blockIdx.x;
  const float* xr = x + (size_t)t * D_;
  float s = 0.f;
  for (int i = threadIdx.x; i < D_; i += 256) s += xr[i];
  red[threadIdx.x] = s;
  __syncthreads();
  for (int o = 128; o > 0; o >>= 1) {
    if (threadIdx.x < o) red[threadIdx.x] += red[threadIdx.x + o];
    __syncthreads();
  }
  const float mu = red[0] / D_;
  __syncthreads();
  float v = 0.f;
  for (int i = threadIdx.x; i < D_; i += 256) { float d = xr[i] - mu; v += d * d; }
  red[threadIdx.x] = v;
  __syncthreads();
  for (int o = 128; o > 0; o >>= 1) {
    if (threadIdx.x < o) red[threadIdx.x] += red[threadIdx.x + o];
    __syncthreads();
  }
  const float rstd = rsqrtf(red[0] / D_ + EPS_);
  for (int i = threadIdx.x; i < D_; i += 256)
    out[(size_t)t * D_ + i] = (bf16)((xr[i] - mu) * rstd * w[i] + b[i]);
}

// ---------------------------------------------------------------- low-rank gate path
__global__ __launch_bounds__(256)
void gate_lowrank(const bf16* __restrict__ xn, const float* __restrict__ Wgk1,
                  const float* __restrict__ Wgk2, const float* __restrict__ bgk,
                  float* __restrict__ gk) {
  __shared__ float red[256];
  __shared__ float lr[LR_];
  const int t = blockIdx.x;
  const bf16* xr = xn + (size_t)t * D_;
  for (int r0 = 0; r0 < LR_; ++r0) {
    float s = 0.f;
    for (int i = threadIdx.x; i < D_; i += 256) s += (float)xr[i] * Wgk1[(size_t)i * LR_ + r0];
    red[threadIdx.x] = s;
    __syncthreads();
    for (int o = 128; o > 0; o >>= 1) {
      if (threadIdx.x < o) red[threadIdx.x] += red[threadIdx.x + o];
      __syncthreads();
    }
    if (threadIdx.x == 0) lr[r0] = red[0];
    __syncthreads();
  }
  const int NK = H_ * DK_;
  for (int j = threadIdx.x; j < NK; j += 256) {
    float s = bgk[j];
    for (int r0 = 0; r0 < LR_; ++r0) s += lr[r0] * Wgk2[(size_t)r0 * NK + j];
    gk[(size_t)t * NK + j] = log_sigmoidf_(s) * (1.0f / GATE_NORM_);
  }
}

// ---------------------------------------------------------------- WMMA GEMM, 128x256x64 tile, TDM-fed
// 8 waves in 2x4; each wave computes 64x64 = 16 accumulators (32 WMMA / k-stage).
// EPI: 0 = store f32; 1 = +resid -> f32; 2 = +bias, GELU -> bf16; 3 = +bias +resid -> f32
template <int EPI>
__global__ __launch_bounds__(256)
void gemm_bf16_wmma(const bf16* __restrict__ A, const bf16* __restrict__ Bt,
                    const float* __restrict__ bias, const float* __restrict__ resid,
                    float* __restrict__ Cf, bf16* __restrict__ Cb, int M, int N, int K) {
  __shared__ __align__(32) bf16 sA[2][128][64];   // M-major [m][k], 32 KB
  __shared__ __align__(32) bf16 sB[2][256][64];   // N-major [n][k], 64 KB
  const int tid = threadIdx.x;
  const int lane = tid & 31, wid = tid >> 5;
  const int wm = wid >> 2;                        // 0..1 -> 64 rows
  const int wn = wid & 3;                         // 0..3 -> 64 cols
  const int half = lane >> 4, r = lane & 15;
  const int m0 = blockIdx.y * 128, n0 = blockIdx.x * 256;

#if USE_TDM
  auto stage = [&](int ks, int buf) {
    if (wid == 0) {
      tdm_load_2d_bf16((unsigned int)(__SIZE_TYPE__)&sA[buf][0][0],
                       A + (size_t)m0 * K + ks * 64, (unsigned int)K, (unsigned int)M,
                       64u, 128u, (unsigned int)K);
      tdm_load_2d_bf16((unsigned int)(__SIZE_TYPE__)&sB[buf][0][0],
                       Bt + (size_t)n0 * K + ks * 64, (unsigned int)K, (unsigned int)N,
                       64u, 256u, (unsigned int)K);
    }
  };
#else
  const int lrow = tid >> 1, lcol = (tid & 1) * 32;
  auto stage = [&](int ks, int buf) {
    const bf16* ga = A + (size_t)(m0 + lrow) * K + ks * 64 + lcol;
    v8bf a0 = *(const v8bf*)ga, a1 = *(const v8bf*)(ga + 8);
    v8bf a2 = *(const v8bf*)(ga + 16), a3 = *(const v8bf*)(ga + 24);
    *(v8bf*)&sA[buf][lrow][lcol] = a0; *(v8bf*)&sA[buf][lrow][lcol + 8] = a1;
    *(v8bf*)&sA[buf][lrow][lcol + 16] = a2; *(v8bf*)&sA[buf][lrow][lcol + 24] = a3;
    for (int rr = 0; rr < 2; ++rr) {
      const int row = lrow + rr * 128;
      const bf16* gb = Bt + (size_t)(n0 + row) * K + ks * 64 + lcol;
      v8bf b0 = *(const v8bf*)gb, b1 = *(const v8bf*)(gb + 8);
      v8bf b2 = *(const v8bf*)(gb + 16), b3 = *(const v8bf*)(gb + 24);
      *(v8bf*)&sB[buf][row][lcol] = b0; *(v8bf*)&sB[buf][row][lcol + 8] = b1;
      *(v8bf*)&sB[buf][row][lcol + 16] = b2; *(v8bf*)&sB[buf][row][lcol + 24] = b3;
    }
  };
#endif

  v8f acc[4][4];
  for (int i = 0; i < 4; ++i)
    for (int j = 0; j < 4; ++j)
      for (int z = 0; z < 8; ++z) acc[i][j][z] = 0.f;

  const int nk = K >> 6;
  stage(0, 0);
  wait_tensorcnt0();
  __syncthreads();
  for (int ks = 0; ks < nk; ++ks) {
    const int cur = ks & 1;
    if (ks + 1 < nk) stage(ks + 1, cur ^ 1);
    for (int ks2 = 0; ks2 < 2; ++ks2) {
      v16bf af[4], bfr[4];
      for (int mt = 0; mt < 4; ++mt)
        af[mt] = load_a_frag(&sA[cur][0][0], 64, wm * 64 + mt * 16 + r, ks2 * 32 + half * 16);
      for (int nt = 0; nt < 4; ++nt)
        bfr[nt] = load_b_frag(&sB[cur][0][0], 64, wn * 64 + nt * 16 + r, ks2 * 32 + half * 16);
      for (int mt = 0; mt < 4; ++mt)
        for (int nt = 0; nt < 4; ++nt)
          acc[mt][nt] = wmma_bf16(af[mt], bfr[nt], acc[mt][nt]);
    }
    wait_tensorcnt0();
    __syncthreads();
  }

  for (int mt = 0; mt < 4; ++mt)
    for (int nt = 0; nt < 4; ++nt)
      for (int r2 = 0; r2 < 8; ++r2) {
        const int row = m0 + wm * 64 + mt * 16 + half * 8 + r2;
        const int col = n0 + wn * 64 + nt * 16 + r;
        const size_t idx = (size_t)row * N + col;
        const float v = acc[mt][nt][r2];
        if (EPI == 0) Cf[idx] = v;
        else if (EPI == 1) Cf[idx] = v + resid[idx];
        else if (EPI == 2) Cb[idx] = (bf16)geluf_(v + bias[col]);
        else Cf[idx] = v + bias[col] + resid[idx];
      }
}

// ---------------------------------------------------------------- GLA chunkwise scan (1 block per (b,h))
__global__ __launch_bounds__(256)
void gla_scan(const float* __restrict__ q, const float* __restrict__ k,
              const float* __restrict__ v, const float* __restrict__ gk,
              float* __restrict__ o) {
  __shared__ __align__(32) float sS[DK_][DV_];    // running state, f32
  __shared__ __align__(32) float sG[CH_][DK_];    // cumsum gates
  __shared__ __align__(32) bf16 sQg[CH_][DK_];
  __shared__ __align__(32) bf16 sKg[CH_][DK_];    // [s][dk] (N-major for A=qg*kg^T)
  __shared__ __align__(32) bf16 sVt[DV_][CH_];    // v transposed [dv][s]
  __shared__ __align__(32) bf16 sSt[DV_][DK_];    // S transposed, bf16 [dv][dk]
  __shared__ __align__(32) bf16 sKeT[DK_][CH_];   // kend transposed [dk][s]
  __shared__ __align__(32) bf16 sA[CH_][CH_];     // masked intra-chunk matrix

  const int tid = threadIdx.x;
  const int lane = tid & 31, wid = tid >> 5;
  const int half = lane >> 4, r = lane & 15;
  const int bh = blockIdx.x;
  const int b = bh >> 4, h = bh & 15;
  const float scale = 0.125f;                     // DK^-0.5
  const size_t qk_row = (size_t)H_ * DK_;         // 1024
  const size_t vg_row = (size_t)H_ * DV_;         // 2048

  for (int i = tid; i < DK_ * DV_; i += 256) sS[i >> 7][i & 127] = 0.f;
  __syncthreads();

  for (int ch = 0; ch < T_ / CH_; ++ch) {
    const int t0 = b * T_ + ch * CH_;
    // prefetch next chunk's rows while this chunk computes
    if (ch + 1 < T_ / CH_ && tid < CH_) {
      const int t1 = t0 + CH_ + tid;
      __builtin_prefetch(&q[(size_t)t1 * qk_row + h * DK_], 0, 1);
      __builtin_prefetch(&k[(size_t)t1 * qk_row + h * DK_], 0, 1);
      __builtin_prefetch(&v[(size_t)t1 * vg_row + h * DV_], 0, 1);
      __builtin_prefetch(&gk[(size_t)t1 * qk_row + h * DK_], 0, 1);
    }
    // 1) cumulative gates: 4 segments of 16 rows + serial fixup
    {
      const int d = tid & 63, qd = tid >> 6;
      float run = 0.f;
      for (int i = 0; i < 16; ++i) {
        run += gk[(size_t)(t0 + qd * 16 + i) * qk_row + h * DK_ + d];
        sG[qd * 16 + i][d] = run;
      }
    }
    __syncthreads();
    if (tid < 64) {
      const int d = tid;
      for (int qd = 1; qd < 4; ++qd) {
        const float off = sG[qd * 16 - 1][d];
        for (int i = 0; i < 16; ++i) sG[qd * 16 + i][d] += off;
      }
    }
    __syncthreads();
    // 2) build bf16 operand tiles
    for (int idx = tid; idx < CH_ * DK_; idx += 256) {
      const int c = idx >> 6, d = idx & 63;
      const float gcd = sG[c][d], gT = sG[CH_ - 1][d];
      const float qv = q[(size_t)(t0 + c) * qk_row + h * DK_ + d];
      const float kv = k[(size_t)(t0 + c) * qk_row + h * DK_ + d];
      sQg[c][d] = (bf16)(qv * scale * __expf(gcd));
      sKg[c][d] = (bf16)(kv * __expf(-gcd));
      sKeT[d][c] = (bf16)(kv * __expf(gT - gcd));
    }
    for (int idx = tid; idx < CH_ * DV_; idx += 256) {
      const int c = idx >> 7, dv = idx & 127;
      sVt[dv][c] = (bf16)v[(size_t)(t0 + c) * vg_row + h * DV_ + dv];
    }
    for (int idx = tid; idx < DK_ * DV_; idx += 256) {
      const int dk = idx >> 7, dv = idx & 127;
      sSt[dv][dk] = (bf16)sS[dk][dv];
    }
    __syncthreads();
    // 3) A = tril(qg @ kg^T): 16 tiles, 2 per wave, K=64
    for (int ti = 0; ti < 2; ++ti) {
      const int tile = wid * 2 + ti;
      const int mt = tile >> 2, nt = tile & 3;
      v8f a; for (int z = 0; z < 8; ++z) a[z] = 0.f;
      for (int ks = 0; ks < 2; ++ks) {
        v16bf af = load_a_frag(&sQg[0][0], DK_, mt * 16 + r, ks * 32 + half * 16);
        v16bf bfr = load_b_frag(&sKg[0][0], DK_, nt * 16 + r, ks * 32 + half * 16);
        a = wmma_bf16(af, bfr, a);
      }
      for (int r2 = 0; r2 < 8; ++r2) {
        const int row = mt * 16 + half * 8 + r2;
        const int col = nt * 16 + r;
        sA[row][col] = (bf16)(col <= row ? a[r2] : 0.f);
      }
    }
    __syncthreads();
    // 4) o = A @ v + qg @ S : 32 tiles (4x8), 4 per wave
    for (int ti = 0; ti < 4; ++ti) {
      const int tile = wid * 4 + ti;
      const int mt = tile >> 3, nt = tile & 7;
      v8f a; for (int z = 0; z < 8; ++z) a[z] = 0.f;
      for (int ks = 0; ks < 2; ++ks) {
        v16bf af = load_a_frag(&sA[0][0], CH_, mt * 16 + r, ks * 32 + half * 16);
        v16bf bfr = load_b_frag(&sVt[0][0], CH_, nt * 16 + r, ks * 32 + half * 16);
        a = wmma_bf16(af, bfr, a);
      }
      for (int ks = 0; ks < 2; ++ks) {
        v16bf af = load_a_frag(&sQg[0][0], DK_, mt * 16 + r, ks * 32 + half * 16);
        v16bf bfr = load_b_frag(&sSt[0][0], DK_, nt * 16 + r, ks * 32 + half * 16);
        a = wmma_bf16(af, bfr, a);
      }
      for (int r2 = 0; r2 < 8; ++r2) {
        const int row = mt * 16 + half * 8 + r2;
        const int col = nt * 16 + r;
        o[(size_t)(t0 + row) * vg_row + h * DV_ + col] = a[r2];
      }
    }
    // 5) S = exp(gT) * S + kend^T @ v : 32 tiles (dk x dv), 4 per wave
    for (int ti = 0; ti < 4; ++ti) {
      const int tile = wid * 4 + ti;
      const int mt = tile >> 3, nt = tile & 7;
      v8f a;
      for (int r2 = 0; r2 < 8; ++r2) {
        const int row = mt * 16 + half * 8 + r2;
        const int col = nt * 16 + r;
        a[r2] = sS[row][col] * __expf(sG[CH_ - 1][row]);
      }
      for (int ks = 0; ks < 2; ++ks) {
        v16bf af = load_a_frag(&sKeT[0][0], CH_, mt * 16 + r, ks * 32 + half * 16);
        v16bf bfr = load_b_frag(&sVt[0][0], CH_, nt * 16 + r, ks * 32 + half * 16);
        a = wmma_bf16(af, bfr, a);
      }
      for (int r2 = 0; r2 < 8; ++r2) {
        const int row = mt * 16 + half * 8 + r2;
        const int col = nt * 16 + r;
        sS[row][col] = a[r2];
      }
    }
    __syncthreads();
  }
}

// ---------------------------------------------------------------- per-head RMS + SiLU gate -> bf16
__global__ __launch_bounds__(256)
void rms_gate(const float* __restrict__ o, const float* __restrict__ g,
              const float* __restrict__ rms_w, bf16* __restrict__ out) {
  __shared__ float sq[256];
  const int t = blockIdx.x;
  const int hh = threadIdx.x >> 4, l16 = threadIdx.x & 15;
  const float* orow = o + (size_t)t * (H_ * DV_) + hh * DV_;
  const float* grow = g + (size_t)t * (H_ * DV_) + hh * DV_;
  float s = 0.f;
  for (int j = l16; j < DV_; j += 16) { const float x = orow[j]; s += x * x; }
  sq[threadIdx.x] = s;
  __syncthreads();
  if (l16 == 0) {
    float tot = 0.f;
    for (int i = 0; i < 16; ++i) tot += sq[hh * 16 + i];
    sq[hh * 16] = rsqrtf(tot / DV_ + EPS_);
  }
  __syncthreads();
  const float rstd = sq[hh * 16];
  bf16* orow_out = out + (size_t)t * (H_ * DV_) + hh * DV_;
  for (int j = l16; j < DV_; j += 16) {
    const float rv = orow[j] * rstd * rms_w[j];
    const float gv = grow[j];
    orow_out[j] = (bf16)(rv * gv * sigmoidf_(gv));
  }
}

// ---------------------------------------------------------------- host
extern "C" void kernel_launch(void* const* d_in, const int* in_sizes, int n_in,
                              void* d_out, int out_size, void* d_ws, size_t ws_size,
                              hipStream_t stream) {
  (void)in_sizes; (void)n_in; (void)out_size; (void)ws_size;
  const float* x     = (const float*)d_in[0];
  const float* ln1_w = (const float*)d_in[1];
  const float* ln1_b = (const float*)d_in[2];
  const float* Wq    = (const float*)d_in[3];
  const float* Wk    = (const float*)d_in[4];
  const float* Wv    = (const float*)d_in[5];
  const float* Wg    = (const float*)d_in[6];
  const float* Wgk1  = (const float*)d_in[7];
  const float* Wgk2  = (const float*)d_in[8];
  const float* bgk   = (const float*)d_in[9];
  const float* rms_w = (const float*)d_in[10];
  const float* Wo    = (const float*)d_in[11];
  const float* ln2_w = (const float*)d_in[12];
  const float* ln2_b = (const float*)d_in[13];
  const float* W1    = (const float*)d_in[14];
  const float* b1    = (const float*)d_in[15];
  const float* W2    = (const float*)d_in[16];
  const float* b2    = (const float*)d_in[17];
  float* out = (float*)d_out;

  size_t off = 0;
  auto alloc = [&](size_t bytes) -> void* {
    void* p = (void*)((char*)d_ws + off);
    off = (off + bytes + 255) & ~(size_t)255;
    return p;
  };
  bf16* xn   = (bf16*)alloc((size_t)M_ * D_ * 2);
  bf16* hbf  = (bf16*)alloc((size_t)M_ * D_ * 2);
  bf16* ogbf = (bf16*)alloc((size_t)M_ * D_ * 2);
  bf16* ubf  = (bf16*)alloc((size_t)M_ * 4 * D_ * 2);
  float* qb  = (float*)alloc((size_t)M_ * H_ * DK_ * 4);
  float* kb  = (float*)alloc((size_t)M_ * H_ * DK_ * 4);
  float* vb  = (float*)alloc((size_t)M_ * H_ * DV_ * 4);
  float* gb  = (float*)alloc((size_t)M_ * H_ * DV_ * 4);
  float* gkb = (float*)alloc((size_t)M_ * H_ * DK_ * 4);
  float* ob  = (float*)alloc((size_t)M_ * H_ * DV_ * 4);
  float* x1  = (float*)alloc((size_t)M_ * D_ * 4);
  bf16* wqT = (bf16*)alloc((size_t)D_ * (H_ * DK_) * 2);
  bf16* wkT = (bf16*)alloc((size_t)D_ * (H_ * DK_) * 2);
  bf16* wvT = (bf16*)alloc((size_t)D_ * (H_ * DV_) * 2);
  bf16* wgT = (bf16*)alloc((size_t)D_ * (H_ * DV_) * 2);
  bf16* woT = (bf16*)alloc((size_t)(H_ * DV_) * D_ * 2);
  bf16* w1T = (bf16*)alloc((size_t)D_ * 4 * D_ * 2);
  bf16* w2T = (bf16*)alloc((size_t)4 * D_ * D_ * 2);

  // weights -> bf16, transposed to N-major for contiguous B-fragments
  cvt_transpose<<<dim3((H_ * DK_) / 32, D_ / 32), 256, 0, stream>>>(Wq, wqT, D_, H_ * DK_);
  cvt_transpose<<<dim3((H_ * DK_) / 32, D_ / 32), 256, 0, stream>>>(Wk, wkT, D_, H_ * DK_);
  cvt_transpose<<<dim3((H_ * DV_) / 32, D_ / 32), 256, 0, stream>>>(Wv, wvT, D_, H_ * DV_);
  cvt_transpose<<<dim3((H_ * DV_) / 32, D_ / 32), 256, 0, stream>>>(Wg, wgT, D_, H_ * DV_);
  cvt_transpose<<<dim3(D_ / 32, (H_ * DV_) / 32), 256, 0, stream>>>(Wo, woT, H_ * DV_, D_);
  cvt_transpose<<<dim3((4 * D_) / 32, D_ / 32), 256, 0, stream>>>(W1, w1T, D_, 4 * D_);
  cvt_transpose<<<dim3(D_ / 32, (4 * D_) / 32), 256, 0, stream>>>(W2, w2T, 4 * D_, D_);

  layernorm_to_bf16<<<M_, 256, 0, stream>>>(x, ln1_w, ln1_b, xn);

  gemm_bf16_wmma<0><<<dim3((H_ * DK_) / 256, M_ / 128), 256, 0, stream>>>(xn, wqT, nullptr, nullptr, qb, nullptr, M_, H_ * DK_, D_);
  gemm_bf16_wmma<0><<<dim3((H_ * DK_) / 256, M_ / 128), 256, 0, stream>>>(xn, wkT, nullptr, nullptr, kb, nullptr, M_, H_ * DK_, D_);
  gemm_bf16_wmma<0><<<dim3((H_ * DV_) / 256, M_ / 128), 256, 0, stream>>>(xn, wvT, nullptr, nullptr, vb, nullptr, M_, H_ * DV_, D_);
  gemm_bf16_wmma<0><<<dim3((H_ * DV_) / 256, M_ / 128), 256, 0, stream>>>(xn, wgT, nullptr, nullptr, gb, nullptr, M_, H_ * DV_, D_);

  gate_lowrank<<<M_, 256, 0, stream>>>(xn, Wgk1, Wgk2, bgk, gkb);

  gla_scan<<<B_ * H_, 256, 0, stream>>>(qb, kb, vb, gkb, ob);

  rms_gate<<<M_, 256, 0, stream>>>(ob, gb, rms_w, ogbf);

  gemm_bf16_wmma<1><<<dim3(D_ / 256, M_ / 128), 256, 0, stream>>>(ogbf, woT, nullptr, x, x1, nullptr, M_, D_, H_ * DV_);

  layernorm_to_bf16<<<M_, 256, 0, stream>>>(x1, ln2_w, ln2_b, hbf);

  gemm_bf16_wmma<2><<<dim3((4 * D_) / 256, M_ / 128), 256, 0, stream>>>(hbf, w1T, b1, nullptr, nullptr, ubf, M_, 4 * D_, D_);

  gemm_bf16_wmma<3><<<dim3(D_ / 256, M_ / 128), 256, 0, stream>>>(ubf, w2T, b2, x1, out, nullptr, M_, D_, 4 * D_);
}
